// selfAttnV1_63118839382024
// MI455X (gfx1250) — matrix operand order repaired and checked by reading, not compiled, not run on hardware
//
#include <hip/hip_runtime.h>
#include <hip/hip_bf16.h>

#define HID 1024
#define SEQ 4096
#define BATCH 4

typedef __attribute__((ext_vector_type(16))) __bf16        v16bf;
typedef __attribute__((ext_vector_type(8)))  float         v8f;
typedef __attribute__((ext_vector_type(4)))  unsigned int  v4u;
typedef __attribute__((ext_vector_type(8)))  unsigned int  v8u;

union Frag {
    v16bf v;
    v4u   q[2];
};

__device__ __forceinline__ unsigned short f2bf(float f) {
    unsigned int x = __float_as_uint(f);
    x += 0x7FFFu + ((x >> 16) & 1u);          // round-to-nearest-even
    return (unsigned short)(x >> 16);
}

// ---------------------------------------------------------------------------
// fp32 -> bf16 conversion (optionally scaled); n4 = elements/4
// ---------------------------------------------------------------------------
__global__ void cvt_f32_bf16(const float* __restrict__ src,
                             unsigned short* __restrict__ dst,
                             int n4, float scale) {
    int i = blockIdx.x * blockDim.x + threadIdx.x;
    if (i < n4) {
        float4 f = ((const float4*)src)[i];
        uint2 o;
        o.x = (unsigned)f2bf(f.x * scale) | ((unsigned)f2bf(f.y * scale) << 16);
        o.y = (unsigned)f2bf(f.z * scale) | ((unsigned)f2bf(f.w * scale) << 16);
        ((uint2*)dst)[i] = o;
    }
}

// ---------------------------------------------------------------------------
// QKV projection: out[m,n] = sum_k Xb[m,k] * Wb[n,k]   (x @ W^T)
// Block: 256 thr = 8 waves; block tile 128(M) x 64(N); wave tile 16x64.
// transposeV==0 -> row-major bf16 out [16384,1024]
// transposeV==1 -> outVT[b, n, s] bf16 (V transposed, packed b128 stores)
// ---------------------------------------------------------------------------
__global__ __launch_bounds__(256) void qkv_gemm(
        const unsigned short* __restrict__ Xb,     // [16384,1024] bf16
        const unsigned short* __restrict__ Wb,     // [1024,1024]  bf16
        unsigned short* __restrict__ outQK,
        unsigned short* __restrict__ outVT,
        int transposeV) {
    const int lane = threadIdx.x & 31;
    const int wave = threadIdx.x >> 5;
    const int lm   = lane & 15;   // M for A / N for B,C
    const int lh   = lane >> 4;   // half-wave select
    const int mbase = blockIdx.x * 128 + wave * 16;
    const int nbase = blockIdx.y * 64;

    v8f c[4] = {};
    const unsigned short* arow = Xb + (size_t)(mbase + lm) * HID;

    for (int k = 0; k < HID; k += 32) {
        Frag a;
        a.q[0] = *(const v4u*)(arow + k + 8 * lh);
        a.q[1] = *(const v4u*)(arow + k + 16 + 8 * lh);
#pragma unroll
        for (int nf = 0; nf < 4; ++nf) {
            const unsigned short* brow =
                Wb + (size_t)(nbase + nf * 16 + lm) * HID + k + 8 * lh;
            Frag b;
            b.q[0] = *(const v4u*)(brow);
            b.q[1] = *(const v4u*)(brow + 16);
            c[nf] = __builtin_amdgcn_wmma_f32_16x16x32_bf16(
                        false, a.v, false, b.v, (short)0, c[nf], false, false);
        }
    }

    if (!transposeV) {
#pragma unroll
        for (int nf = 0; nf < 4; ++nf) {
            int ncol = nbase + nf * 16 + lm;
#pragma unroll
            for (int i = 0; i < 8; ++i) {
                int m = mbase + 8 * lh + i;
                outQK[(size_t)m * HID + ncol] = f2bf(c[nf][i]);
            }
        }
    } else {
        int bidx = mbase / SEQ;
        int s0   = (mbase % SEQ) + 8 * lh;     // 8 consecutive s per lane
#pragma unroll
        for (int nf = 0; nf < 4; ++nf) {
            int ncol = nbase + nf * 16 + lm;
            v4u pk;
            pk.x = (unsigned)f2bf(c[nf][0]) | ((unsigned)f2bf(c[nf][1]) << 16);
            pk.y = (unsigned)f2bf(c[nf][2]) | ((unsigned)f2bf(c[nf][3]) << 16);
            pk.z = (unsigned)f2bf(c[nf][4]) | ((unsigned)f2bf(c[nf][5]) << 16);
            pk.w = (unsigned)f2bf(c[nf][6]) | ((unsigned)f2bf(c[nf][7]) << 16);
            *(v4u*)(outVT + ((size_t)bidx * HID + ncol) * SEQ + s0) = pk;
        }
    }
}

// ---------------------------------------------------------------------------
// Attention: one block (8 waves) per 16-query tile.
// LDS: scores f32 [16][4096] (256KB, aliased later as P bf16) + Q bf16 (32KB)
//      + stats (4KB).  Total 299008 B  (<= 320KB WGP LDS).
// ---------------------------------------------------------------------------
__global__ __launch_bounds__(256) void attn_kernel(
        const unsigned short* __restrict__ Qs,   // [16384,1024] bf16 (pre-scaled 1/32)
        const unsigned short* __restrict__ Kb,   // [16384,1024] bf16
        const unsigned short* __restrict__ VT,   // [B,1024,4096] bf16
        float* __restrict__ Out) {               // [16384,1024] f32
    extern __shared__ char smem[];
    float*          sS = (float*)smem;                          // [16][4096] f32
    unsigned short* sP = (unsigned short*)smem;                 // [16][4096] bf16 (alias)
    unsigned short* sQ = (unsigned short*)(smem + 262144);      // [16][1024]
    float*          sStat  = (float*)(smem + 262144 + 32768);
    float*          rowmax = sStat;                             // 16
    float*          rowsum = sStat + 16;                        // 16
    float*          red    = sStat + 32;                        // 256

    const int tid  = threadIdx.x;
    const int lane = tid & 31;
    const int wave = tid >> 5;
    const int lm   = lane & 15;
    const int lh   = lane >> 4;
    const int qb   = blockIdx.x * 16;          // global token row base
    const int bidx = qb / SEQ;

    // ---- phase 0: stage Q tile into LDS via Tensor Data Mover (TDM) ----
    // 16 rows x 1024 bf16 are contiguous in Qs -> one 1-D 32KB tile,
    // data_size=8B, tensor_dim0=tile_dim0=4096 elements.
    if (wave == 0) {
        unsigned int       lq = (unsigned int)(size_t)sQ;                 // LDS byte addr
        unsigned long long ga = (unsigned long long)(size_t)(Qs + (size_t)qb * HID);
        v4u g0;
        g0.x = 1u;                                     // count=1 (valid), user mode
        g0.y = lq;                                     // lds_addr
        g0.z = (unsigned int)ga;                       // global_addr[31:0]
        g0.w = (unsigned int)(ga >> 32) | (2u << 30);  // global_addr[56:32] | type=2
        v8u g1;
        g1[0] = 3u << 16;       // data_size = 8 bytes, no mask/flags
        g1[1] = 4096u << 16;    // tensor_dim0 = 4096 (8B units), bits[63:48]
        g1[2] = 1u << 16;       // tensor_dim1 = 1
        g1[3] = 4096u << 16;    // tile_dim0 = 4096, bits[127:112]
        g1[4] = 0u;             // tile_dim1 = tile_dim2 = 0 (1-D tile)
        g1[5] = 4096u;          // tensor_dim0_stride[31:0]
        g1[6] = 0u;
        g1[7] = 0u;
        asm volatile("tensor_load_to_lds %0, %1" :: "s"(g0), "s"(g1) : "memory");
        __builtin_amdgcn_s_wait_tensorcnt(0);
    }
    __syncthreads();

    // ---- phase 1: S = Qs @ K^T ; wave w handles keys [w*512, w*512+512) ----
    {
        const int kb0 = wave * 512;
        const unsigned short* aro = sQ + lm * HID;
        for (int nf = 0; nf < 32; ++nf) {
            v8f c = {};
            const int key = kb0 + nf * 16 + lm;
            const unsigned short* brow = Kb + ((size_t)bidx * SEQ + key) * HID;
            for (int k = 0; k < HID; k += 32) {
                Frag a, b;
                a.q[0] = *(const v4u*)(aro + k + 8 * lh);
                a.q[1] = *(const v4u*)(aro + k + 16 + 8 * lh);
                b.q[0] = *(const v4u*)(brow + k + 8 * lh);
                b.q[1] = *(const v4u*)(brow + k + 16 + 8 * lh);
                c = __builtin_amdgcn_wmma_f32_16x16x32_bf16(
                        false, a.v, false, b.v, (short)0, c, false, false);
            }
#pragma unroll
            for (int i = 0; i < 8; ++i)
                sS[(8 * lh + i) * SEQ + key] = c[i];
        }
    }
    __syncthreads();

    // ---- phase 2a: row max (deterministic tree reduce, 16 thr/row) ----
    {
        const int row = tid >> 4;
        const int c0  = (tid & 15) * 256;
        float m = -3.0e38f;
        for (int j = 0; j < 256; ++j) m = fmaxf(m, sS[row * SEQ + c0 + j]);
        red[tid] = m;
    }
    __syncthreads();
    if (tid < 16) {
        float m = red[tid * 16];
        for (int j = 1; j < 16; ++j) m = fmaxf(m, red[tid * 16 + j]);
        rowmax[tid] = m;
    }
    __syncthreads();

    // ---- phase 2b: row sum of exp (deterministic) ----
    {
        const int row = tid >> 4;
        const int c0  = (tid & 15) * 256;
        const float mx = rowmax[row];
        float s = 0.f;
        for (int j = 0; j < 256; ++j) s += __expf(sS[row * SEQ + c0 + j] - mx);
        red[tid] = s;
    }
    __syncthreads();
    if (tid < 16) {
        float s = 0.f;
        for (int j = 0; j < 16; ++j) s += red[tid * 16 + j];
        rowsum[tid] = s;
    }
    __syncthreads();

    // ---- phase 2c: in-place convert P = exp(S-max) -> bf16 (alias-safe:
    //      pass p reads f32 [p,p+2048); bf16 writes land in f32 [p/2,p/2+1024),
    //      always fully read in earlier passes or before the intra-pass barrier)
    for (int pass = 0; pass < 16 * SEQ; pass += 2048) {
        const int idx = pass + tid * 8;
        const float mx = rowmax[idx >> 12];
        float v[8];
#pragma unroll
        for (int j = 0; j < 8; ++j) v[j] = __expf(sS[idx + j] - mx);
        __syncthreads();
        v4u pk;
        pk.x = (unsigned)f2bf(v[0]) | ((unsigned)f2bf(v[1]) << 16);
        pk.y = (unsigned)f2bf(v[2]) | ((unsigned)f2bf(v[3]) << 16);
        pk.z = (unsigned)f2bf(v[4]) | ((unsigned)f2bf(v[5]) << 16);
        pk.w = (unsigned)f2bf(v[6]) | ((unsigned)f2bf(v[7]) << 16);
        *(v4u*)(sP + idx) = pk;
        __syncthreads();
    }

    // ---- phase 3: O = P @ V ; wave w handles output cols [w*128, w*128+128) ----
    {
        v8f o[8] = {};
        const int nb0 = wave * 128;
        const unsigned short* aro = sP + lm * SEQ;
        for (int jb = 0; jb < SEQ; jb += 32) {
            Frag a;
            a.q[0] = *(const v4u*)(aro + jb + 8 * lh);
            a.q[1] = *(const v4u*)(aro + jb + 16 + 8 * lh);
#pragma unroll
            for (int nf = 0; nf < 8; ++nf) {
                const int n = nb0 + nf * 16 + lm;
                const unsigned short* brow =
                    VT + ((size_t)bidx * HID + n) * SEQ + jb + 8 * lh;
                Frag b;
                b.q[0] = *(const v4u*)(brow);
                b.q[1] = *(const v4u*)(brow + 16);
                o[nf] = __builtin_amdgcn_wmma_f32_16x16x32_bf16(
                            false, a.v, false, b.v, (short)0, o[nf], false, false);
            }
        }
#pragma unroll
        for (int nf = 0; nf < 8; ++nf) {
            const int n = nb0 + nf * 16 + lm;
#pragma unroll
            for (int i = 0; i < 8; ++i) {
                const int r = 8 * lh + i;
                Out[((size_t)qb + r) * HID + n] = o[nf][i] / rowsum[r];
            }
        }
    }
}

// ---------------------------------------------------------------------------
extern "C" void kernel_launch(void* const* d_in, const int* in_sizes, int n_in,
                              void* d_out, int out_size, void* d_ws, size_t ws_size,
                              hipStream_t stream) {
    const float* X  = (const float*)d_in[0];
    const float* Wq = (const float*)d_in[1];
    const float* Wk = (const float*)d_in[2];
    const float* Wv = (const float*)d_in[3];
    float* Out = (float*)d_out;

    char* ws = (char*)d_ws;
    const size_t SZ_X = (size_t)BATCH * SEQ * HID * 2;   // 32 MB bf16
    const size_t SZ_W = (size_t)HID * HID * 2;           //  2 MB bf16
    unsigned short* Xb  = (unsigned short*)(ws);
    unsigned short* Wqb = (unsigned short*)(ws + SZ_X);
    unsigned short* Wkb = (unsigned short*)(ws + SZ_X + SZ_W);
    unsigned short* Wvb = (unsigned short*)(ws + SZ_X + 2 * SZ_W);
    unsigned short* Qs  = (unsigned short*)(ws + SZ_X + 3 * SZ_W);
    unsigned short* Kb  = (unsigned short*)(ws + 2 * SZ_X + 3 * SZ_W);
    unsigned short* VT  = (unsigned short*)(ws + 3 * SZ_X + 3 * SZ_W);

    const int nX = BATCH * SEQ * HID;   // 16,777,216
    const int nW = HID * HID;           //  1,048,576
    cvt_f32_bf16<<<nX / 4 / 256, 256, 0, stream>>>(X,  Xb,  nX / 4, 1.0f);
    cvt_f32_bf16<<<nW / 4 / 256, 256, 0, stream>>>(Wq, Wqb, nW / 4, 0.03125f); // fold 1/sqrt(1024)
    cvt_f32_bf16<<<nW / 4 / 256, 256, 0, stream>>>(Wk, Wkb, nW / 4, 1.0f);
    cvt_f32_bf16<<<nW / 4 / 256, 256, 0, stream>>>(Wv, Wvb, nW / 4, 1.0f);

    dim3 g1(BATCH * SEQ / 128, HID / 64);
    qkv_gemm<<<g1, 256, 0, stream>>>(Xb, Wqb, Qs, nullptr, 0);
    qkv_gemm<<<g1, 256, 0, stream>>>(Xb, Wkb, Kb, nullptr, 0);
    qkv_gemm<<<g1, 256, 0, stream>>>(Xb, Wvb, nullptr, VT, 1);

    const size_t shmem = 262144 + 32768 + 4096;  // 299008 B <= 320KB WGP LDS
    attn_kernel<<<BATCH * SEQ / 16, 256, shmem, stream>>>(Qs, Kb, VT, Out);
}
